// PolicyNet_7636451852644
// MI455X (gfx1250) — compile-verified, matrix operand order checked by reading, not run on hardware
//
#include <hip/hip_runtime.h>
#include <math.h>

typedef __attribute__((ext_vector_type(16))) _Float16 v16h;
typedef __attribute__((ext_vector_type(8)))  _Float16 v8h;
typedef __attribute__((ext_vector_type(8)))  float    v8f;

#define FEAT  256
#define EPS   1e-5f
#define PIXT  64          // pixels per workgroup
#define HSTR  264         // padded LDS row stride in halves (528B -> conflict-friendly)
#define LTOT  21760
#define KSAMP 100
#define BATCH 8

// ---------------------------------------------------------------- W1 -> f16
__global__ void cvt_w1(const float* __restrict__ W1, _Float16* __restrict__ W1h) {
    int i = blockIdx.x * blockDim.x + threadIdx.x;      // 65536 elements
    W1h[i] = (_Float16)W1[i];
}

__global__ void zero_buf(float* __restrict__ p, int n) {
    int i = blockIdx.x * blockDim.x + threadIdx.x;
    if (i < n) p[i] = 0.f;
}

// --------------------------------------------------- GN1 stats per (lvl,b,g)
__global__ void gn1_stats(const float* __restrict__ x, int L, int lvl,
                          float* __restrict__ mean, float* __restrict__ rstd) {
    int g = blockIdx.x, b = blockIdx.y;
    const float* base = x + ((size_t)b * FEAT + (size_t)g * 32) * L;  // contiguous 32*L block
    int n = 32 * L;
    float s = 0.f, ss = 0.f;
    for (int i = threadIdx.x; i < n; i += blockDim.x) {
        __builtin_prefetch(base + i + 8 * 256, 0, 1);   // global_prefetch_b8: warm L2 ahead
        float v = base[i]; s += v; ss += v * v;
    }
    __shared__ float rs[256], rss[256];
    rs[threadIdx.x] = s; rss[threadIdx.x] = ss;
    __syncthreads();
    for (int o = 128; o > 0; o >>= 1) {
        if (threadIdx.x < o) {
            rs[threadIdx.x]  += rs[threadIdx.x + o];
            rss[threadIdx.x] += rss[threadIdx.x + o];
        }
        __syncthreads();
    }
    if (threadIdx.x == 0) {
        float mu  = rs[0] / n;
        float var = rss[0] / n - mu * mu;
        int sidx = (lvl * 8 + b) * 8 + g;
        mean[sidx] = mu;
        rstd[sidx] = rsqrtf(var + EPS);
    }
}

// ------------------------------------------------------------ GN2 finalize
__global__ void gn2_fin(const float* __restrict__ s, const float* __restrict__ q,
                        float* __restrict__ m2, float* __restrict__ r2) {
    int i = threadIdx.x;                 // 256 (lvl,b,g) entries
    int lvl = i >> 6;
    const int Ls[4] = {16384, 4096, 1024, 256};
    float n  = 32.f * (float)Ls[lvl];
    float mu = s[i] / n;
    float var = q[i] / n - mu * mu;
    m2[i] = mu;
    r2[i] = rsqrtf(var + EPS);
}

// -------------------------------------------------- fused GN1+GEMM (+stats | +logits)
// PASS 0: y = W1 @ relu(gn1(x)) + b1 ; accumulate GN2 sum/sumsq (atomics)
// PASS 1: recompute y ; logits = W2 @ relu(gn2(y)) + b2
// Each wave owns 2 out-tiles; per out-tile it runs 4 INDEPENDENT accumulator
// chains (one per 16-pixel tile) sharing one A fragment per K-step.
template<int PASS>
__global__ __launch_bounds__(256) void head_gemm(
    const float* __restrict__ x, int L, int lvl,
    const _Float16* __restrict__ W1h,
    const float* __restrict__ g1s, const float* __restrict__ g1b,
    const float* __restrict__ b1,
    const float* __restrict__ g2s, const float* __restrict__ g2b,
    const float* __restrict__ W2,  const float* __restrict__ b2,
    const float* __restrict__ m1,  const float* __restrict__ r1,
    float* __restrict__ gsum, float* __restrict__ gssq,
    const float* __restrict__ m2,  const float* __restrict__ r2,
    float* __restrict__ logits)    // already offset to this level's column base
{
    __shared__ _Float16 hbuf[PIXT * HSTR];   // h tile, [pixel][channel], f16
    __shared__ float    pixAcc[PIXT];

    int b    = blockIdx.y;
    int pix0 = blockIdx.x * PIXT;
    int tid  = threadIdx.x;

    if (PASS == 1 && tid < PIXT) pixAcc[tid] = 0.f;

    // ---- stage h = relu(gn1(x)) into LDS (coalesced 64-float rows) ----
    int l  = tid & 63;           // pixel within tile
    int co = tid >> 6;           // 0..3 channel offset
    for (int cb = 0; cb < FEAT; cb += 4) {
        int c    = cb + co;
        int sidx = (lvl * 8 + b) * 8 + (c >> 5);
        float v  = x[((size_t)b * FEAT + c) * L + pix0 + l];
        float hn = fmaxf((v - m1[sidx]) * r1[sidx] * g1s[c] + g1b[c], 0.f);
        hbuf[l * HSTR + c] = (_Float16)hn;
    }
    __syncthreads();

    // ---- WMMA: 8 waves x 2 out-tiles x 4 pixel-tile chains, K = 8 x 32 ----
    int wave = tid >> 5, lane = tid & 31;
    int mrow = lane & 15;
    int hi   = (lane >> 4) << 3;           // K half-select: 0 (lanes 0-15) / 8 (16-31)
    for (int otl = 0; otl < 2; ++otl) {
        int ot = wave * 2 + otl;
        const _Float16* arow = W1h + (size_t)(ot * 16 + mrow) * FEAT;   // A: W1[o][k]
        v8f acc[4];
        #pragma unroll
        for (int p = 0; p < 4; ++p) acc[p] = (v8f){};
        #pragma unroll
        for (int kb = 0; kb < FEAT; kb += 32) {
            v8h alo = *(const v8h*)(arow + kb + hi);
            v8h ahi = *(const v8h*)(arow + kb + 16 + hi);
            v16h av = __builtin_shufflevector(alo, ahi,
                        0,1,2,3,4,5,6,7,8,9,10,11,12,13,14,15);
            #pragma unroll
            for (int p = 0; p < 4; ++p) {                // 4 independent chains
                const _Float16* brow = hbuf + (p * 16 + mrow) * HSTR;
                v8h blo = *(const v8h*)(brow + kb + hi);
                v8h bhi = *(const v8h*)(brow + kb + 16 + hi);
                v16h bv = __builtin_shufflevector(blo, bhi,
                            0,1,2,3,4,5,6,7,8,9,10,11,12,13,14,15);
                acc[p] = __builtin_amdgcn_wmma_f32_16x16x32_f16(
                    false, av, false, bv, (short)0, acc[p], false, false);
            }
        }
        int obase = ot * 16 + hi;            // C frag: M = r (+8 for lanes>=16)
        int sidx2 = (lvl * 8 + b) * 8 + (ot >> 1);   // all 4 tiles: same GN2 group
        if (PASS == 0) {
            float s = 0.f, ss = 0.f;
            #pragma unroll
            for (int p = 0; p < 4; ++p)
                #pragma unroll
                for (int r = 0; r < 8; ++r) {
                    float y = acc[p][r] + b1[obase + r];
                    s += y; ss += y * y;
                }
            #pragma unroll
            for (int o = 16; o > 0; o >>= 1) {
                s  += __shfl_xor(s,  o, 32);
                ss += __shfl_xor(ss, o, 32);
            }
            if (lane == 0) {
                atomicAdd(&gsum[sidx2], s);
                atomicAdd(&gssq[sidx2], ss);
            }
        } else {
            float mu = m2[sidx2], rs = r2[sidx2];
            #pragma unroll
            for (int p = 0; p < 4; ++p) {
                float pp = 0.f;
                #pragma unroll
                for (int r = 0; r < 8; ++r) {
                    int   o = obase + r;
                    float y = acc[p][r] + b1[o];
                    float z = fmaxf((y - mu) * rs * g2s[o] + g2b[o], 0.f);
                    pp += z * W2[o];
                }
                pp += __shfl_xor(pp, 16, 32);  // combine M 0-7 with M 8-15 (same pixel)
                if (lane < 16) atomicAdd(&pixAcc[p * 16 + lane], pp);  // ds_add_f32
            }
        }
    }
    if (PASS == 1) {
        __syncthreads();
        if (tid < PIXT)
            logits[(size_t)b * LTOT + pix0 + tid] = pixAcc[tid] + b2[0];
    }
}

// ------------------------------------------------- top-100 per batch row
__global__ void topk_kernel(const float* __restrict__ logits, int* __restrict__ ids) {
    __shared__ unsigned int mask[LTOT / 32];   // exclusion bitmask, 2720B
    __shared__ float bv[256];
    __shared__ int   bi[256];
    int b = blockIdx.x, tid = threadIdx.x;
    for (int i = tid; i < LTOT / 32; i += blockDim.x) mask[i] = 0u;
    __syncthreads();
    const float* lg = logits + (size_t)b * LTOT;
    for (int k = 0; k < KSAMP; ++k) {
        float best = -INFINITY; int idx = 0x7FFFFFFF;
        for (int i = tid; i < LTOT; i += blockDim.x) {
            if (mask[i >> 5] & (1u << (i & 31))) continue;
            float v = lg[i];
            if (v > best || (v == best && i < idx)) { best = v; idx = i; }
        }
        bv[tid] = best; bi[tid] = idx;
        __syncthreads();
        for (int o = 128; o > 0; o >>= 1) {
            if (tid < o) {
                float v2 = bv[tid + o]; int i2 = bi[tid + o];
                if (v2 > bv[tid] || (v2 == bv[tid] && i2 < bi[tid])) {
                    bv[tid] = v2; bi[tid] = i2;
                }
            }
            __syncthreads();
        }
        if (tid == 0) {
            ids[b * KSAMP + k] = bi[0];
            mask[bi[0] >> 5] |= 1u << (bi[0] & 31);
        }
        __syncthreads();
    }
}

// --------------------------------------------------------------- launcher
extern "C" void kernel_launch(void* const* d_in, const int* in_sizes, int n_in,
                              void* d_out, int out_size, void* d_ws, size_t ws_size,
                              hipStream_t stream) {
    const float* xs[4] = {(const float*)d_in[0], (const float*)d_in[1],
                          (const float*)d_in[2], (const float*)d_in[3]};
    const float* g1s = (const float*)d_in[4];
    const float* g1b = (const float*)d_in[5];
    const float* W1  = (const float*)d_in[6];
    const float* b1  = (const float*)d_in[7];
    const float* g2s = (const float*)d_in[8];
    const float* g2b = (const float*)d_in[9];
    const float* W2  = (const float*)d_in[10];
    const float* b2  = (const float*)d_in[11];

    // workspace layout: W1 f16 (128KB) + 6 x 256-float stats arrays
    _Float16* W1h = (_Float16*)d_ws;
    float* m1   = (float*)((char*)d_ws + (size_t)FEAT * FEAT * sizeof(_Float16));
    float* r1   = m1 + 256;
    float* gsum = r1 + 256;
    float* gssq = gsum + 256;
    float* m2   = gssq + 256;
    float* r2   = m2 + 256;

    int*   ids    = (int*)d_out;                       // [8,100] int32
    float* logits = (float*)d_out + BATCH * KSAMP;     // [8,21760] f32

    const int Ls[4]   = {16384, 4096, 1024, 256};
    const int offs[4] = {0, 16384, 20480, 21504};

    cvt_w1<<<FEAT * FEAT / 256, 256, 0, stream>>>(W1, W1h);
    zero_buf<<<2, 256, 0, stream>>>(gsum, 512);        // zero gsum+gssq
    for (int lvl = 0; lvl < 4; ++lvl)
        gn1_stats<<<dim3(8, BATCH), 256, 0, stream>>>(xs[lvl], Ls[lvl], lvl, m1, r1);
    for (int lvl = 0; lvl < 4; ++lvl)
        head_gemm<0><<<dim3(Ls[lvl] / PIXT, BATCH), 256, 0, stream>>>(
            xs[lvl], Ls[lvl], lvl, W1h, g1s, g1b, b1, g2s, g2b, W2, b2,
            m1, r1, gsum, gssq, m2, r2, nullptr);
    gn2_fin<<<1, 256, 0, stream>>>(gsum, gssq, m2, r2);
    for (int lvl = 0; lvl < 4; ++lvl)
        head_gemm<1><<<dim3(Ls[lvl] / PIXT, BATCH), 256, 0, stream>>>(
            xs[lvl], Ls[lvl], lvl, W1h, g1s, g1b, b1, g2s, g2b, W2, b2,
            m1, r1, gsum, gssq, m2, r2, logits + offs[lvl]);
    topk_kernel<<<BATCH, 256, 0, stream>>>(logits, ids);
}
